// GemmaAttention_89043261980901
// MI455X (gfx1250) — matrix operand order repaired and checked by reading, not compiled, not run on hardware
//
#include <hip/hip_runtime.h>
#include <hip/hip_bf16.h>
#include <stdint.h>

// ---------------- problem constants ----------------
#define TLEN   4096
#define HID    3072
#define NH     16
#define NKV    8
#define HD     256
#define QSZ    (NH * HD)          // 4096
#define KVSZ   (NKV * HD)         // 2048
#define NQKV   (QSZ + 2 * KVSZ)   // 8192
#define SCALE  0.0625f            // HD^-0.5
#define TPAD   56                 // padded k-stride (elements): 112B, 16B-aligned, conflict-free

typedef __attribute__((ext_vector_type(16))) __bf16 v16bf;
typedef __attribute__((ext_vector_type(8)))  float  v8f;

// ---------------- helpers ----------------
__device__ __forceinline__ unsigned short f2bf(float f) {
    union { float f; unsigned u; } c; c.f = f;
    unsigned u = c.u;
    unsigned r = u + 0x7fffu + ((u >> 16) & 1u);   // RNE
    return (unsigned short)(r >> 16);
}
__device__ __forceinline__ float bf2f(unsigned short h) {
    union { unsigned u; float f; } c; c.u = ((unsigned)h) << 16;
    return c.f;
}
// build a 16-element bf16 fragment from two 16B chunks
__device__ __forceinline__ v16bf ld_frag(const unsigned short* p0, const unsigned short* p1) {
    union { uint4 q[2]; v16bf v; } u;
    u.q[0] = *(const uint4*)p0;
    u.q[1] = *(const uint4*)p1;
    return u.v;
}
__device__ __forceinline__ v8f wmma_bf(v16bf a, v16bf b, v8f c) {
    return __builtin_amdgcn_wmma_f32_16x16x32_bf16(false, a, false, b, (short)0, c, false, false);
}

// ---- CDNA5 async global->LDS (ASYNCcnt path), inline asm: portable on both toolchains
__device__ __forceinline__ unsigned lds_addr(const void* p) {
    return (unsigned)(uintptr_t)p;                 // low 32 bits of generic addr = LDS offset
}
__device__ __forceinline__ void async_b128(unsigned lds, const unsigned short* g) {
    asm volatile("global_load_async_to_lds_b128 %0, %1, off"
                 :: "v"(lds), "v"(g) : "memory");
}
template<int N>
__device__ __forceinline__ void wait_async() {
    asm volatile("s_wait_asynccnt %0" :: "i"(N) : "memory");
}

// ---------------- pack: f32 -> bf16 elementwise ----------------
__global__ void cvt_f32_bf16(const float* __restrict__ src, unsigned short* __restrict__ dst, int n) {
    int i = blockIdx.x * blockDim.x + threadIdx.x;
    if (i < n) dst[i] = f2bf(src[i]);
}

// ---------------- pack: transpose f32 [K][N] -> bf16 [N][K] ----------------
__global__ __launch_bounds__(256) void trans_pack(const float* __restrict__ src,
                                                  unsigned short* __restrict__ dst,
                                                  int K, int N) {
    __shared__ float tile[32][33];
    int n0 = blockIdx.x << 5, k0 = blockIdx.y << 5;
    int tx = threadIdx.x, ty = threadIdx.y;
    for (int r = ty; r < 32; r += 8)
        tile[r][tx] = src[(size_t)(k0 + r) * N + n0 + tx];
    __syncthreads();
    for (int r = ty; r < 32; r += 8)
        dst[(size_t)(n0 + r) * K + k0 + tx] = f2bf(tile[tx][r]);
}

// ---------------- tiled GEMM core: 128x128 block tile, 8 waves of 32x64 ----------------
// A bf16 [M][K] row-major, BT bf16 [N][K]. Both staged in double-buffered LDS
// via global_load_async_to_lds_b128 (ASYNCcnt). Per 32-k-step: 16KB in, 8 WMMAs/wave.
__device__ __forceinline__ void issue_chunk(const unsigned short* Ag, const unsigned short* Bg,
                                            int ldK, int k0, int mbase, int nbase, int tid,
                                            unsigned short* sAbuf, unsigned short* sBbuf) {
#pragma unroll
    for (int i = 0; i < 2; ++i) {
        int idx = tid * 2 + i;              // 512 16B chunks per 128x32 tile
        int row = idx >> 2, c4 = idx & 3;   // c4: 8-element chunk within k-step
        async_b128(lds_addr(sAbuf + row * TPAD + c4 * 8),
                   Ag + (size_t)(mbase + row) * ldK + k0 + c4 * 8);
        async_b128(lds_addr(sBbuf + row * TPAD + c4 * 8),
                   Bg + (size_t)(nbase + row) * ldK + k0 + c4 * 8);
    }
}

__device__ __forceinline__ void gemm_core(const unsigned short* Ag, const unsigned short* Bg,
                                          int K, int mbase, int nbase,
                                          unsigned short* sA, unsigned short* sB,  // [2][128*TPAD]
                                          v8f acc[2][4]) {
    const int tid  = threadIdx.x;
    const int lane = tid & 31;
    const int w    = tid >> 5;
    const int mw   = w & 3;                 // 4 waves down  (M)
    const int nw   = w >> 2;                // 2 waves across(N)
    const int lo = lane & 15, hi = lane >> 4;
    const int ka  = hi << 3;                // A frag: +0 / +8
    const int kbo = hi << 3;                // B frag: +0 / +8

    const int NCH = K >> 5;
    issue_chunk(Ag, Bg, K, 0, mbase, nbase, tid, sA, sB);

    for (int c = 0; c < NCH; ++c) {
        const int cur = (c & 1) * (128 * TPAD);
        if (c + 1 < NCH) {
            const int nxt = ((c + 1) & 1) * (128 * TPAD);
            issue_chunk(Ag, Bg, K, (c + 1) << 5, mbase, nbase, tid, sA + nxt, sB + nxt);
            wait_async<4>();                // loads retire in order: chunk c complete
        } else {
            wait_async<0>();
        }
        __syncthreads();

        const unsigned short* sAc = sA + cur;
        const unsigned short* sBc = sB + cur;
        v16bf a[2];
#pragma unroll
        for (int h = 0; h < 2; ++h) {
            const unsigned short* p = sAc + (mw * 32 + h * 16 + lo) * TPAD + ka;
            a[h] = ld_frag(p, p + 16);
        }
#pragma unroll
        for (int j = 0; j < 4; ++j) {
            const unsigned short* p = sBc + (nw * 64 + j * 16 + lo) * TPAD + kbo;
            v16bf b = ld_frag(p, p + 16);
            acc[0][j] = wmma_bf(a[0], b, acc[0][j]);
            acc[1][j] = wmma_bf(a[1], b, acc[1][j]);
        }
        __syncthreads();                    // protect buffer being refilled next iteration
    }
}

// ---------------- QKV GEMM with routed epilogue ----------------
__global__ __launch_bounds__(256) void qkv_gemm(const unsigned short* __restrict__ A,
                                                const unsigned short* __restrict__ BT,
                                                unsigned short* __restrict__ qb,
                                                unsigned short* __restrict__ kb,
                                                unsigned short* __restrict__ vtb) {
    __shared__ unsigned short sA[2 * 128 * TPAD];
    __shared__ unsigned short sB[2 * 128 * TPAD];
    const int mbase = blockIdx.y << 7;
    const int nbase = blockIdx.x << 7;
    v8f acc[2][4] = {};
    gemm_core(A, BT, HID, mbase, nbase, sA, sB, acc);

    const int lane = threadIdx.x & 31;
    const int w = threadIdx.x >> 5, mw = w & 3, nw = w >> 2;
    const int lo = lane & 15, mb = (lane >> 4) << 3;
#pragma unroll
    for (int h = 0; h < 2; ++h)
#pragma unroll
        for (int j = 0; j < 4; ++j) {
            int col = nbase + nw * 64 + j * 16 + lo;
#pragma unroll
            for (int i = 0; i < 8; ++i) {
                int t = mbase + mw * 32 + h * 16 + mb + i;
                unsigned short v = f2bf(acc[h][j][i]);
                if (col < QSZ)             qb[(size_t)t * QSZ + col] = v;
                else if (col < QSZ + KVSZ) kb[(size_t)t * KVSZ + (col - QSZ)] = v;
                else                       vtb[(size_t)(col - QSZ - KVSZ) * TLEN + t] = v;
            }
        }
}

// ---------------- output GEMM: [T,QSZ] x [QSZ,HID] -> f32 out ----------------
__global__ __launch_bounds__(256) void out_gemm(const unsigned short* __restrict__ A,
                                                const unsigned short* __restrict__ BT,
                                                float* __restrict__ out) {
    __shared__ unsigned short sA[2 * 128 * TPAD];
    __shared__ unsigned short sB[2 * 128 * TPAD];
    const int mbase = blockIdx.y << 7;
    const int nbase = blockIdx.x << 7;
    v8f acc[2][4] = {};
    gemm_core(A, BT, QSZ, mbase, nbase, sA, sB, acc);

    const int lane = threadIdx.x & 31;
    const int w = threadIdx.x >> 5, mw = w & 3, nw = w >> 2;
    const int lo = lane & 15, mb = (lane >> 4) << 3;
#pragma unroll
    for (int h = 0; h < 2; ++h)
#pragma unroll
        for (int j = 0; j < 4; ++j) {
            int col = nbase + nw * 64 + j * 16 + lo;
#pragma unroll
            for (int i = 0; i < 8; ++i)
                out[(size_t)(mbase + mw * 32 + h * 16 + mb + i) * HID + col] = acc[h][j][i];
        }
}

// ---------------- RoPE in-place on bf16 q,k ----------------
__global__ void rope_kernel(const int* __restrict__ pos,
                            unsigned short* __restrict__ qb,
                            unsigned short* __restrict__ kb) {
    int tid = blockIdx.x * blockDim.x + threadIdx.x;   // T * 24 * 128 threads
    int j = tid & 127;
    int r = tid >> 7;
    int hh = r % (NH + NKV);
    int t  = r / (NH + NKV);
    if (t >= TLEN) return;
    float invf = __expf(-logf(10000.f) * (float)j * (1.f / 128.f));
    float ang  = (float)pos[t] * invf;
    float s, c;
    __sincosf(ang, &s, &c);
    unsigned short* p = (hh < NH) ? (qb + (size_t)t * QSZ + hh * HD)
                                  : (kb + (size_t)t * KVSZ + (hh - NH) * HD);
    float x1 = bf2f(p[j]), x2 = bf2f(p[j + 128]);
    p[j]       = f2bf(x1 * c - x2 * s);
    p[j + 128] = f2bf(x2 * c + x1 * s);
}

// ---------------- flash attention: 1 wave per (head, 16-query tile) ----------------
__global__ __launch_bounds__(32) void attn_kernel(const unsigned short* __restrict__ qb,
                                                  const unsigned short* __restrict__ kb,
                                                  const unsigned short* __restrict__ vtb,
                                                  unsigned short* __restrict__ ob) {
    __shared__ unsigned short sp[16 * 32];             // P tile staging (C-layout -> A-layout)
    const int lane = threadIdx.x & 31;
    const int head = blockIdx.x >> 8;
    const int qt0  = (blockIdx.x & 255) << 4;
    const int kvh  = head >> 1;                        // G = 2
    const int lo = lane & 15, hi = lane >> 4;

    v16bf qf[8];
    {
        const unsigned short* qrow = qb + (size_t)(qt0 + lo) * QSZ + head * HD;
#pragma unroll
        for (int kk = 0; kk < 8; ++kk) {
            int off = kk * 32 + (hi << 3);
            qf[kk] = ld_frag(qrow + off, qrow + off + 16);
        }
    }
    float mr[8], lr[8];
#pragma unroll
    for (int i = 0; i < 8; ++i) { mr[i] = -3.0e38f; lr[i] = 0.f; }
    v8f O[16] = {};

    const int qlast = qt0 + 15;
    for (int s0 = 0; s0 <= qlast; s0 += 32) {
        v8f S0 = {}, S1 = {};
#pragma unroll
        for (int kk = 0; kk < 8; ++kk) {
            int kof = kk * 32 + (hi << 3);
            const unsigned short* b0 = kb + (size_t)(s0 + lo) * KVSZ + kvh * HD + kof;
            const unsigned short* b1 = kb + (size_t)(s0 + 16 + lo) * KVSZ + kvh * HD + kof;
            S0 = wmma_bf(qf[kk], ld_frag(b0, b0 + 16), S0);
            S1 = wmma_bf(qf[kk], ld_frag(b1, b1 + 16), S1);
        }
        float alpha[8];
#pragma unroll
        for (int i = 0; i < 8; ++i) {
            int t  = qt0 + (hi << 3) + i;
            int sA_ = s0 + lo, sB_ = sA_ + 16;
            float a = S0[i] * SCALE;
            float b = S1[i] * SCALE;
            a = (sA_ > t) ? -3.0e38f : a;
            b = (sB_ > t) ? -3.0e38f : b;
            float mx = fmaxf(a, b);
#pragma unroll
            for (int off = 1; off < 16; off <<= 1) mx = fmaxf(mx, __shfl_xor(mx, off, 32));
            float mn = fmaxf(mr[i], mx);
            float al = __expf(mr[i] - mn);
            float p0 = __expf(a - mn);
            float p1 = __expf(b - mn);
            float rs = p0 + p1;
#pragma unroll
            for (int off = 1; off < 16; off <<= 1) rs += __shfl_xor(rs, off, 32);
            lr[i] = lr[i] * al + rs;
            mr[i] = mn;
            alpha[i] = al;
            int m = (hi << 3) + i;
            sp[m * 32 + lo]      = f2bf(p0);
            sp[m * 32 + 16 + lo] = f2bf(p1);
        }
#pragma unroll
        for (int d = 0; d < 16; ++d)
#pragma unroll
            for (int i = 0; i < 8; ++i) O[d][i] *= alpha[i];

        __asm__ volatile("s_wait_dscnt 0" ::: "memory");   // single-wave LDS RAW
        v16bf pf;
        {
            const unsigned short* pr = &sp[lo * 32 + (hi << 3)];
            pf = ld_frag(pr, pr + 16);
        }
#pragma unroll
        for (int d = 0; d < 16; ++d) {
            const unsigned short* vc = vtb + (size_t)(kvh * HD + d * 16 + lo) * TLEN + s0 + (hi << 3);
            O[d] = wmma_bf(pf, ld_frag(vc, vc + 16), O[d]);
        }
    }
    float inv[8];
#pragma unroll
    for (int i = 0; i < 8; ++i) inv[i] = 1.f / lr[i];
#pragma unroll
    for (int d = 0; d < 16; ++d)
#pragma unroll
        for (int i = 0; i < 8; ++i) {
            int t = qt0 + (hi << 3) + i;
            ob[(size_t)t * QSZ + head * HD + d * 16 + lo] = f2bf(O[d][i] * inv[i]);
        }
}

// ---------------- host launcher ----------------
extern "C" void kernel_launch(void* const* d_in, const int* in_sizes, int n_in,
                              void* d_out, int out_size, void* d_ws, size_t ws_size,
                              hipStream_t stream) {
    const int*   positions = (const int*)d_in[0];
    const float* hidden    = (const float*)d_in[1];
    const float* w_qkv     = (const float*)d_in[2];
    const float* w_o       = (const float*)d_in[3];
    float* out = (float*)d_out;

    char* ws = (char*)d_ws;
    unsigned short* hsb   = (unsigned short*)(ws);                       // [T][HID]      24 MB
    unsigned short* wqkvT = (unsigned short*)(ws + 25165824);            // [NQKV][HID]   48 MB
    unsigned short* woT   = (unsigned short*)(ws + 75497472);            // [HID][QSZ]    24 MB
    unsigned short* qb    = (unsigned short*)(ws + 100663296);           // [T][QSZ]      32 MB
    unsigned short* kbuf  = (unsigned short*)(ws + 134217728);           // [T][KVSZ]     16 MB
    unsigned short* vtb   = (unsigned short*)(ws + 150994944);           // [KVSZ][T]     16 MB
    unsigned short* ob    = (unsigned short*)(ws + 167772160);           // [T][QSZ]      32 MB

    {
        int n = TLEN * HID;
        cvt_f32_bf16<<<(n + 255) / 256, 256, 0, stream>>>(hidden, hsb, n);
    }
    trans_pack<<<dim3(NQKV / 32, HID / 32), dim3(32, 8), 0, stream>>>(w_qkv, wqkvT, HID, NQKV);
    trans_pack<<<dim3(HID / 32, QSZ / 32), dim3(32, 8), 0, stream>>>(w_o, woT, QSZ, HID);

    // QKV GEMM: 128x128 block tiles -> grid (8192/128, 4096/128)
    qkv_gemm<<<dim3(NQKV / 128, TLEN / 128), 256, 0, stream>>>(hsb, wqkvT, qb, kbuf, vtb);

    {
        int n = TLEN * (NH + NKV) * 128;
        rope_kernel<<<(n + 255) / 256, 256, 0, stream>>>(positions, qb, kbuf);
    }

    attn_kernel<<<NH * (TLEN / 16), 32, 0, stream>>>(qb, kbuf, vtb, ob);

    // output GEMM: grid (3072/128, 4096/128)
    out_gemm<<<dim3(HID / 128, TLEN / 128), 256, 0, stream>>>(ob, woT, out);
}
